// Transformer_35527969472532
// MI455X (gfx1250) — matrix operand order repaired and checked by reading, not compiled, not run on hardware
//
#include <hip/hip_runtime.h>

// ---------------------------------------------------------------------------
// Transformer encoder forward (6 layers) for MI455X / gfx1250, wave32 + WMMA.
// GEMMs: v_wmma_f32_16x16x32_bf16, LDS double-buffered via gfx1250 async
// global->LDS copies (ASYNCcnt) when available, else pipelined reg staging.
// ---------------------------------------------------------------------------

typedef unsigned short u16;
typedef __attribute__((ext_vector_type(16))) __bf16 bf16x16;
typedef __attribute__((ext_vector_type(8)))  float  f32x8;
typedef __attribute__((ext_vector_type(4)))  int    int4v;

#define T_SEQ 256
#define BATCH 32
#define NF    1024
#define NH    16
#define HDIM  64
#define PDIM  1024
#define FFD   4096
#define NLAY  6
#define M_TOK (T_SEQ * BATCH)   // 8192 tokens

#if defined(__HIP_DEVICE_COMPILE__) && defined(__gfx1250__) && \
    __has_builtin(__builtin_amdgcn_global_load_async_to_lds_b128) && \
    __has_builtin(__builtin_amdgcn_s_wait_asynccnt)
#define HAVE_ASYNC_LDS 1
#else
#define HAVE_ASYNC_LDS 0
#endif

__device__ __forceinline__ u16 f32_to_bf16(float f) {
    unsigned u = __float_as_uint(f);
    u += 0x7FFFu + ((u >> 16) & 1u);       // round-to-nearest-even
    return (u16)(u >> 16);
}
__device__ __forceinline__ float bf16_to_f32(u16 h) {
    return __uint_as_float(((unsigned)h) << 16);
}

#if HAVE_ASYNC_LDS
// Builtin signature (probe-confirmed by diagnostics): (v4i AS1*, v4i AS3*, imm, imm)
typedef __attribute__((address_space(1))) int4v* as1_v4i;
typedef __attribute__((address_space(3))) int4v* as3_v4i;
// 16B per lane, global -> LDS through the async engine (tracked by ASYNCcnt).
__device__ __forceinline__ void async_cp16(const u16* g, const u16* l) {
    __builtin_amdgcn_global_load_async_to_lds_b128(
        (as1_v4i)(unsigned long long)g,
        (as3_v4i)(unsigned)(unsigned long long)l, 0, 0);
}
#endif

// ---------------------------------------------------------------------------
// fp32 -> bf16 cast (vectorized x4)
// ---------------------------------------------------------------------------
__global__ void cast_f32_to_bf16(const float* __restrict__ in,
                                 u16* __restrict__ out, long n4) {
    long i = (long)blockIdx.x * blockDim.x + threadIdx.x;
    if (i >= n4) return;
    float4 v = ((const float4*)in)[i];
    ushort4 r;
    r.x = f32_to_bf16(v.x);
    r.y = f32_to_bf16(v.y);
    r.z = f32_to_bf16(v.z);
    r.w = f32_to_bf16(v.w);
    ((ushort4*)out)[i] = r;
}

// ---------------------------------------------------------------------------
// LayerNorm over last dim (N=1024). One wave32 per row, 8 rows per block.
// ---------------------------------------------------------------------------
template <bool OUT_BF16>
__global__ __launch_bounds__(256) void ln_kernel(const float* __restrict__ x,
                                                 const float* __restrict__ g,
                                                 const float* __restrict__ b,
                                                 void* __restrict__ out, int rows) {
    int row  = blockIdx.x * 8 + (threadIdx.x >> 5);
    int lane = threadIdx.x & 31;
    if (row >= rows) return;
    const float* xr = x + (long)row * NF;
    float vals[32];
    float s = 0.f, ss = 0.f;
    #pragma unroll
    for (int j = 0; j < 32; ++j) {
        float v = xr[lane + 32 * j];
        vals[j] = v;
        s += v;
        ss += v * v;
    }
    #pragma unroll
    for (int off = 16; off > 0; off >>= 1) {
        s  += __shfl_xor(s,  off, 32);
        ss += __shfl_xor(ss, off, 32);
    }
    float mean = s * (1.0f / NF);
    float var  = ss * (1.0f / NF) - mean * mean;
    float inv  = rsqrtf(var + 1e-5f);
    #pragma unroll
    for (int j = 0; j < 32; ++j) {
        int c = lane + 32 * j;
        float y = (vals[j] - mean) * inv * g[c] + b[c];
        if (OUT_BF16)
            ((u16*)out)[(long)row * NF + c] = f32_to_bf16(y);
        else
            ((float*)out)[(long)row * NF + c] = y;
    }
}

// ---------------------------------------------------------------------------
// Generic bf16 WMMA GEMM:  C[m,p] = sum_k A[m,k] * Bw[p,k]  (both K-major)
// Optional epilogue: +bias[p], ReLU, +residual (fp32), bf16 or fp32 output.
// Block tile BM x BP, BK=32. Wave tile 32x64 = 2x4 WMMA 16x16x32 tiles.
// LDS is double-buffered; tiles stream in via async global->LDS copies.
// ---------------------------------------------------------------------------
template <int BM, int BP, int WM, int WN,
          bool HAS_BIAS, bool DO_RELU, bool RESID, bool BF16_OUT>
__global__ __launch_bounds__(WM * WN * 32) void gemm_bf16_kernel(
    const u16* __restrict__ A, long azs,
    const u16* __restrict__ Bw, long bzs,
    const float* __restrict__ bias,
    const float* __restrict__ Cin,
    void* __restrict__ Cout, long czs,
    int K, int lda, int ldb, int ldc) {
    constexpr int BK    = 32;
    constexpr int LDS_W = BK + 8;                // padded LDS row (elements)
    constexpr int NTHR  = WM * WN * 32;
    constexpr int ACH   = (BM * BK) / (NTHR * 8);  // 16B chunks/thread for A
    constexpr int BCH   = (BP * BK) / (NTHR * 8);  // 16B chunks/thread for B
    constexpr int NCH   = ACH + BCH;               // async ops per tile/thread

    __shared__ u16 As[2 * BM * LDS_W];
    __shared__ u16 Bs[2 * BP * LDS_W];

    const int z = blockIdx.z;
    A  += (long)z * azs;
    Bw += (long)z * bzs;
    const long cbase   = (long)z * czs;
    const int  rowBase = blockIdx.y * BM;
    const int  colBase = blockIdx.x * BP;

    const int tid     = threadIdx.x;
    const int lane    = tid & 31;
    const int wave    = tid >> 5;
    const int wm      = wave / WN;
    const int wn      = wave % WN;
    const int halfsel = lane >> 4;   // 0: lanes 0-15, 1: lanes 16-31
    const int l16     = lane & 15;

    // Per-thread copy descriptors (computed once; no muls in the K loop).
    const u16* gA[ACH]; int lA[ACH];
    #pragma unroll
    for (int i = 0; i < ACH; ++i) {
        int e = tid + i * NTHR;              // 16B chunk id; 4 chunks per row
        int r = e >> 2, c = (e & 3) * 8;
        gA[i] = A + (long)(rowBase + r) * lda + c;
        lA[i] = r * LDS_W + c;
    }
    const u16* gB[BCH]; int lB[BCH];
    #pragma unroll
    for (int i = 0; i < BCH; ++i) {
        int e = tid + i * NTHR;
        int r = e >> 2, c = (e & 3) * 8;
        gB[i] = Bw + (long)(colBase + r) * ldb + c;
        lB[i] = r * LDS_W + c;
    }

    f32x8 acc[2][4];
    #pragma unroll
    for (int rt = 0; rt < 2; ++rt)
        #pragma unroll
        for (int ct = 0; ct < 4; ++ct)
            #pragma unroll
            for (int v = 0; v < 8; ++v)
                acc[rt][ct][v] = 0.0f;

    const int KT = K / BK;

    // Consume one resident LDS tile: load fragments in the CDNA5 wave32 WMMA
    // layouts and issue the 2x4 WMMA block.
    auto compute_tile = [&](int buf) {
        const u16* Ab = &As[buf * BM * LDS_W];
        const u16* Bb = &Bs[buf * BP * LDS_W];
        bf16x16 afrag[2];
        #pragma unroll
        for (int rt = 0; rt < 2; ++rt) {
            int r = wm * 32 + rt * 16 + l16;
            const u16* p = &Ab[r * LDS_W + halfsel * 8];
            *(uint4*)&afrag[rt]         = *(const uint4*)p;        // K 0-7 / 8-15
            *(((uint4*)&afrag[rt]) + 1) = *(const uint4*)(p + 16); // K 16-23 / 24-31
        }
        bf16x16 bfrag[4];
        #pragma unroll
        for (int ct = 0; ct < 4; ++ct) {
            int cc = wn * 64 + ct * 16 + l16;
            const u16* p = &Bb[cc * LDS_W + halfsel * 8];
            *(uint4*)&bfrag[ct]         = *(const uint4*)p;        // K 0-7 / 8-15
            *(((uint4*)&bfrag[ct]) + 1) = *(const uint4*)(p + 16); // K 16-23 / 24-31
        }
        #pragma unroll
        for (int rt = 0; rt < 2; ++rt)
            #pragma unroll
            for (int ct = 0; ct < 4; ++ct)
                acc[rt][ct] = __builtin_amdgcn_wmma_f32_16x16x32_bf16(
                    false, afrag[rt], false, bfrag[ct],
                    (short)0, acc[rt][ct], false, false);
    };

#if HAVE_ASYNC_LDS
    // ---- async double-buffered pipeline (gfx1250 ASYNCcnt path) ----
    auto stage = [&](int buf, int k0) {
        #pragma unroll
        for (int i = 0; i < ACH; ++i)
            async_cp16(gA[i] + k0, &As[buf * BM * LDS_W + lA[i]]);
        #pragma unroll
        for (int i = 0; i < BCH; ++i)
            async_cp16(gB[i] + k0, &Bs[buf * BP * LDS_W + lB[i]]);
    };
    stage(0, 0);
    for (int kt = 0; kt < KT; ++kt) {
        int cur = kt & 1;
        if (kt + 1 < KT) {
            // previous compute on buffer cur^1 finished (barrier at loop tail)
            stage(cur ^ 1, (kt + 1) * BK);
            __builtin_amdgcn_s_wait_asynccnt(NCH);  // tile kt resident (in-order)
        } else {
            __builtin_amdgcn_s_wait_asynccnt(0);
        }
        __syncthreads();            // tile kt resident for every wave
        compute_tile(cur);
        __syncthreads();            // all readers done before buffer reuse
    }
#else
    // ---- fallback: software-pipelined register staging, single buffer ----
    uint4 ra[ACH], rb[BCH];
    #pragma unroll
    for (int i = 0; i < ACH; ++i) ra[i] = *(const uint4*)(gA[i]);
    #pragma unroll
    for (int i = 0; i < BCH; ++i) rb[i] = *(const uint4*)(gB[i]);
    for (int kt = 0; kt < KT; ++kt) {
        __syncthreads();            // readers of previous tile done
        #pragma unroll
        for (int i = 0; i < ACH; ++i) *(uint4*)&As[lA[i]] = ra[i];
        #pragma unroll
        for (int i = 0; i < BCH; ++i) *(uint4*)&Bs[lB[i]] = rb[i];
        __syncthreads();
        if (kt + 1 < KT) {          // next-tile loads overlap current compute
            int k0n = (kt + 1) * BK;
            #pragma unroll
            for (int i = 0; i < ACH; ++i) ra[i] = *(const uint4*)(gA[i] + k0n);
            #pragma unroll
            for (int i = 0; i < BCH; ++i) rb[i] = *(const uint4*)(gB[i] + k0n);
        }
        compute_tile(0);
    }
#endif

    // ---- epilogue: C/D layout = 8 rows per lane, half-wave row offset 8 ----
    #pragma unroll
    for (int rt = 0; rt < 2; ++rt) {
        int r0 = rowBase + wm * 32 + rt * 16 + halfsel * 8;
        #pragma unroll
        for (int ct = 0; ct < 4; ++ct) {
            int c = colBase + wn * 64 + ct * 16 + l16;
            float bv = HAS_BIAS ? bias[c] : 0.0f;
            #pragma unroll
            for (int v = 0; v < 8; ++v) {
                long idx = cbase + (long)(r0 + v) * ldc + c;
                float val = acc[rt][ct][v] + bv;
                if (DO_RELU) val = fmaxf(val, 0.0f);
                if (RESID)   val += Cin[idx];
                if (BF16_OUT) ((u16*)Cout)[idx] = f32_to_bf16(val);
                else          ((float*)Cout)[idx] = val;
            }
        }
    }
}

// ---------------------------------------------------------------------------
// QKV reshape: qkv[t*B+b, 3*PD] -> q,k [B,H,T,HD] (q scaled 1/sqrt(HD)),
//              vt [B,H,HD,T] (V transposed so P*V is also an A.B^T GEMM).
// ---------------------------------------------------------------------------
__global__ __launch_bounds__(256) void qkv_reshape_kernel(
    const u16* __restrict__ qkv, u16* __restrict__ q,
    u16* __restrict__ k, u16* __restrict__ vt) {
    long i = (long)blockIdx.x * 256 + threadIdx.x;   // [b][h][t][d] linear
    int d = (int)(i & 63);
    int t = (int)((i >> 6) & 255);
    int h = (int)((i >> 14) & 15);
    int b = (int)(i >> 18);
    long srow = ((long)t * BATCH + b) * (3 * PDIM) + h * HDIM + d;
    q[i] = f32_to_bf16(bf16_to_f32(qkv[srow]) * 0.125f);   // 1/sqrt(64)
    k[i] = qkv[srow + PDIM];
    vt[(((long)b * NH + h) * HDIM + d) * T_SEQ + t] = qkv[srow + 2 * PDIM];
}

// ---------------------------------------------------------------------------
// In-place softmax over rows of s [Z*T, T] (bf16) with additive mask [T,T].
// One wave32 per row (T=256 -> 8 elements/lane), 8 rows per block.
// ---------------------------------------------------------------------------
__global__ __launch_bounds__(256) void softmax_kernel(u16* __restrict__ s,
                                                      const float* __restrict__ mask) {
    long row  = (long)blockIdx.x * 8 + (threadIdx.x >> 5);
    int  lane = threadIdx.x & 31;
    int  t    = (int)(row & (T_SEQ - 1));
    u16* sr = s + row * T_SEQ;
    const float* mr = mask + (long)t * T_SEQ;
    float v[8];
    float mx = -3.0e38f;
    #pragma unroll
    for (int j = 0; j < 8; ++j) {
        int c = lane + 32 * j;
        float f = bf16_to_f32(sr[c]) + mr[c];
        v[j] = f;
        mx = fmaxf(mx, f);
    }
    #pragma unroll
    for (int off = 16; off > 0; off >>= 1) mx = fmaxf(mx, __shfl_xor(mx, off, 32));
    float sum = 0.f;
    #pragma unroll
    for (int j = 0; j < 8; ++j) { v[j] = __expf(v[j] - mx); sum += v[j]; }
    #pragma unroll
    for (int off = 16; off > 0; off >>= 1) sum += __shfl_xor(sum, off, 32);
    float inv = 1.0f / sum;
    #pragma unroll
    for (int j = 0; j < 8; ++j) sr[lane + 32 * j] = f32_to_bf16(v[j] * inv);
}

// ---------------------------------------------------------------------------
// O transpose: ot [B,H,T,HD] -> o [t*B+b, PD]
// ---------------------------------------------------------------------------
__global__ __launch_bounds__(256) void transpose_o_kernel(const u16* __restrict__ ot,
                                                          u16* __restrict__ o) {
    long i = (long)blockIdx.x * 256 + threadIdx.x;
    int d = (int)(i & 63);
    int t = (int)((i >> 6) & 255);
    int h = (int)((i >> 14) & 15);
    int b = (int)(i >> 18);
    o[((long)t * BATCH + b) * PDIM + h * HDIM + d] = ot[i];
}

// ---------------------------------------------------------------------------
// Host orchestration
// ---------------------------------------------------------------------------
extern "C" void kernel_launch(void* const* d_in, const int* in_sizes, int n_in,
                              void* d_out, int out_size, void* d_ws, size_t ws_size,
                              hipStream_t stream) {
    (void)in_sizes; (void)n_in; (void)out_size; (void)ws_size;

    const float* src       = (const float*)d_in[0];
    const float* attn_mask = (const float*)d_in[1];
    const float* Wqkv      = (const float*)d_in[2];
    const float* bqkv      = (const float*)d_in[3];
    const float* Wo        = (const float*)d_in[4];
    const float* bo        = (const float*)d_in[5];
    const float* ln1_g     = (const float*)d_in[6];
    const float* ln1_b     = (const float*)d_in[7];
    const float* ln2_g     = (const float*)d_in[8];
    const float* ln2_b     = (const float*)d_in[9];
    const float* W1        = (const float*)d_in[10];
    const float* b1        = (const float*)d_in[11];
    const float* W2        = (const float*)d_in[12];
    const float* b2        = (const float*)d_in[13];
    const float* lnf_g     = (const float*)d_in[14];
    const float* lnf_b     = (const float*)d_in[15];

    char*  ws  = (char*)d_ws;
    size_t off = 0;
    auto walloc = [&](size_t bytes) -> void* {
        void* p = ws + off;
        off = (off + bytes + 255) & ~(size_t)255;
        return p;
    };
    const int ZBH = BATCH * NH;   // 512 attention batches
    float* x   = (float*)walloc((size_t)M_TOK * NF * 4);
    u16*   h   = (u16*)walloc((size_t)M_TOK * NF * 2);
    u16*   qkv = (u16*)walloc((size_t)M_TOK * 3 * PDIM * 2);
    u16*   q   = (u16*)walloc((size_t)ZBH * T_SEQ * HDIM * 2);
    u16*   k   = (u16*)walloc((size_t)ZBH * T_SEQ * HDIM * 2);
    u16*   vt  = (u16*)walloc((size_t)ZBH * T_SEQ * HDIM * 2);
    u16*   s   = (u16*)walloc((size_t)ZBH * T_SEQ * T_SEQ * 2);
    u16*   ot  = (u16*)walloc((size_t)ZBH * T_SEQ * HDIM * 2);
    u16*   o   = (u16*)walloc((size_t)M_TOK * PDIM * 2);
    u16*   ff  = (u16*)walloc((size_t)M_TOK * FFD * 2);
    u16* wqkv_b = (u16*)walloc((size_t)3 * PDIM * NF * 2);
    u16* wo_b   = (u16*)walloc((size_t)NF * PDIM * 2);
    u16* w1_b   = (u16*)walloc((size_t)FFD * NF * 2);
    u16* w2_b   = (u16*)walloc((size_t)NF * FFD * 2);

    // residual stream seed
    (void)hipMemcpyAsync(x, src, (size_t)M_TOK * NF * 4, hipMemcpyDeviceToDevice, stream);

    auto cast = [&](const float* in, u16* outp, size_t n) {
        long n4 = (long)(n / 4);
        int blocks = (int)((n4 + 255) / 256);
        cast_f32_to_bf16<<<blocks, 256, 0, stream>>>(in, outp, n4);
    };

    for (int l = 0; l < NLAY; ++l) {
        cast(Wqkv + (size_t)l * 3 * PDIM * NF, wqkv_b, (size_t)3 * PDIM * NF);
        cast(Wo   + (size_t)l * NF * PDIM,     wo_b,   (size_t)NF * PDIM);
        cast(W1   + (size_t)l * FFD * NF,      w1_b,   (size_t)FFD * NF);
        cast(W2   + (size_t)l * NF * FFD,      w2_b,   (size_t)NF * FFD);

        // --- attention block ---
        ln_kernel<true><<<M_TOK / 8, 256, 0, stream>>>(
            x, ln1_g + (size_t)l * NF, ln1_b + (size_t)l * NF, h, M_TOK);

        // qkv = h @ Wqkv^T + bqkv                 [8192 x 3072], K=1024
        gemm_bf16_kernel<128, 128, 4, 2, true, false, false, true>
            <<<dim3(3 * PDIM / 128, M_TOK / 128, 1), 256, 0, stream>>>(
                h, 0, wqkv_b, 0, bqkv + (size_t)l * 3 * PDIM, nullptr,
                qkv, 0, NF, NF, NF, 3 * PDIM);

        qkv_reshape_kernel<<<(ZBH * T_SEQ * HDIM) / 256, 256, 0, stream>>>(qkv, q, k, vt);

        // scores = q @ k^T   per (b,h)            [256 x 256], K=64, Z=512
        gemm_bf16_kernel<128, 128, 4, 2, false, false, false, true>
            <<<dim3(T_SEQ / 128, T_SEQ / 128, ZBH), 256, 0, stream>>>(
                q, (long)T_SEQ * HDIM, k, (long)T_SEQ * HDIM, nullptr, nullptr,
                s, (long)T_SEQ * T_SEQ, HDIM, HDIM, HDIM, T_SEQ);

        softmax_kernel<<<(ZBH * T_SEQ) / 8, 256, 0, stream>>>(s, attn_mask);

        // ot = probs @ V  (V^T stored)            [256 x 64], K=256, Z=512
        gemm_bf16_kernel<128, 64, 4, 1, false, false, false, true>
            <<<dim3(HDIM / 64, T_SEQ / 128, ZBH), 128, 0, stream>>>(
                s, (long)T_SEQ * T_SEQ, vt, (long)T_SEQ * HDIM, nullptr, nullptr,
                ot, (long)T_SEQ * HDIM, T_SEQ, T_SEQ, T_SEQ, HDIM);

        transpose_o_kernel<<<(ZBH * T_SEQ * HDIM) / 256, 256, 0, stream>>>(ot, o);

        // x += o @ Wo^T + bo                      [8192 x 1024], K=1024
        gemm_bf16_kernel<128, 128, 4, 2, true, false, true, false>
            <<<dim3(NF / 128, M_TOK / 128, 1), 256, 0, stream>>>(
                o, 0, wo_b, 0, bo + (size_t)l * NF, x, x, 0, PDIM, PDIM, PDIM, NF);

        // --- FFN block ---
        ln_kernel<true><<<M_TOK / 8, 256, 0, stream>>>(
            x, ln2_g + (size_t)l * NF, ln2_b + (size_t)l * NF, h, M_TOK);

        // ff = relu(h @ W1^T + b1)                [8192 x 4096], K=1024
        gemm_bf16_kernel<128, 128, 4, 2, true, true, false, true>
            <<<dim3(FFD / 128, M_TOK / 128, 1), 256, 0, stream>>>(
                h, 0, w1_b, 0, b1 + (size_t)l * FFD, nullptr, ff, 0, NF, NF, NF, FFD);

        // x += ff @ W2^T + b2                     [8192 x 1024], K=4096
        gemm_bf16_kernel<128, 128, 4, 2, true, false, true, false>
            <<<dim3(NF / 128, M_TOK / 128, 1), 256, 0, stream>>>(
                ff, 0, w2_b, 0, b2 + (size_t)l * NF, x, x, 0, FFD, FFD, FFD, NF);
    }

    // final LayerNorm -> fp32 output
    ln_kernel<false><<<M_TOK / 8, 256, 0, stream>>>(x, lnf_g, lnf_b, d_out, M_TOK);
}
